// Attention_20315195310563
// MI455X (gfx1250) — compile-verified, matrix operand order checked by reading, not run on hardware
//
#include <hip/hip_runtime.h>
#include <hip/hip_bf16.h>

// Problem constants (match reference)
#define DIMC   512
#define NHEADS 8
#define HDIM   64
#define BATCH  4
#define SEQ    2048
#define QKSCALE 0.125f   // 64^-0.5

typedef __attribute__((ext_vector_type(16))) _Float16 v16h;
typedef __attribute__((ext_vector_type(8)))  _Float16 v8h;
typedef __attribute__((ext_vector_type(8)))  float    v8f;
typedef int v4i __attribute__((vector_size(16)));   // matches builtin's pointee

// ---- CDNA5 async global->LDS staging (guarded; sync fallback) ----
#if defined(__gfx1250__) &&                                                   \
    __has_builtin(__builtin_amdgcn_global_load_async_to_lds_b128) &&          \
    __has_builtin(__builtin_amdgcn_s_wait_asynccnt)
#define USE_ASYNC_LDS 1
#define AS1G __attribute__((address_space(1)))
#define AS3L __attribute__((address_space(3)))
static __device__ __forceinline__ AS3L v4i* to_lds(void* p) {
  // AS(3) pointers are the low 32 bits of the generic LDS address
  return (AS3L v4i*)(unsigned int)(unsigned long long)p;
}
static __device__ __forceinline__ AS1G v4i* to_glb(const void* p) {
  return (AS1G v4i*)(unsigned long long)p;
}
#endif

// ---- WMMA helper: D = A(16x32 f16) * B(32x16 f16) + C(16x16 f32) ----
static __device__ __forceinline__ v8f wmma32(v16h a, v16h b, v8f c) {
  return __builtin_amdgcn_wmma_f32_16x16x32_f16(
      /*neg_a=*/false, a, /*neg_b=*/false, b,
      /*c_mod=*/(short)0, c, /*reuse_a=*/false, /*reuse_b=*/false);
}

// A/B fragment loader from a row-major f16 matrix.
// Lane l holds row (caller passes it), K-chunks: lanes 0-15 -> K {0..7,16..23},
// lanes 16-31 -> K {8..15,24..31} relative to k0. (ISA 7.12.2, 16-bit A 16x32)
template <typename PTR>
static __device__ __forceinline__ v16h load_frag_h(PTR base, int row, int ld,
                                                   int k0, int lane) {
  const int hk = (lane & 16) ? 8 : 0;
  PTR p = base + (size_t)row * ld + k0 + hk;
  v8h lo = *(const v8h*)p;
  v8h hi = *(const v8h*)(p + 16);
  v16h r;
#pragma unroll
  for (int e = 0; e < 8; ++e) { r[e] = lo[e]; r[8 + e] = hi[e]; }
  return r;
}

// Same fragment, but source is fp32 row-major (convert to f16 on load).
static __device__ __forceinline__ v16h load_frag_f(const float* __restrict__ base,
                                                   long row, int ld, int k0,
                                                   int lane) {
  const int hk = (lane & 16) ? 8 : 0;
  const float* p = base + (size_t)row * ld + k0 + hk;
  v16h r;
#pragma unroll
  for (int e = 0; e < 8; ++e) {
    r[e]     = (_Float16)p[e];
    r[8 + e] = (_Float16)p[16 + e];
  }
  return r;
}

// ============================================================
// Kernel 1: fused QKV projection  y = x @ qkv_w^T  (no bias),
// scatter into Q (pre-scaled), K, V  f16  [B, H, N, 64].
// One wave per 16x64 output tile (A fragment reused across 4 B tiles).
// M tiles = 8192/16 = 512, N64 tiles = 1536/64 = 24 -> 12288 waves.
// ============================================================
__global__ __launch_bounds__(256) void attn_qkv_kernel(
    const float* __restrict__ x, const float* __restrict__ qkv_w,
    _Float16* __restrict__ Qh, _Float16* __restrict__ Kh,
    _Float16* __restrict__ Vh) {
  const int gwave = (int)((blockIdx.x * blockDim.x + threadIdx.x) >> 5);
  const int lane  = (int)(threadIdx.x & 31);
  const int mt = gwave / 24;
  const int nt = gwave % 24;
  const int m0 = mt * 16;
  const int n0 = nt * 64;
  const int l16 = lane & 15;

  v8f acc[4];
#pragma unroll
  for (int t = 0; t < 4; ++t)
#pragma unroll
    for (int i = 0; i < 8; ++i) acc[t][i] = 0.0f;

#pragma unroll 2
  for (int k0 = 0; k0 < DIMC; k0 += 32) {
    v16h a = load_frag_f(x, (long)(m0 + l16), DIMC, k0, lane);
#pragma unroll
    for (int t = 0; t < 4; ++t) {
      v16h b = load_frag_f(qkv_w, (long)(n0 + t * 16 + l16), DIMC, k0, lane);
      acc[t] = wmma32(a, b, acc[t]);
    }
  }

  // Scatter C tiles (lane = col l16; vgpr r = row r or r+8)
  const int rbase = (lane & 16) ? 8 : 0;
#pragma unroll
  for (int t = 0; t < 4; ++t) {
    const int col = n0 + t * 16 + l16;
    const int which = col >> 9;          // 0=Q 1=K 2=V
    const int h = (col >> 6) & 7;
    const int d = col & 63;
    _Float16* dst = (which == 0) ? Qh : (which == 1) ? Kh : Vh;
    const float sc = (which == 0) ? QKSCALE : 1.0f;
#pragma unroll
    for (int r = 0; r < 8; ++r) {
      const long grow = m0 + rbase + r;      // 0..8191
      const long bb = grow >> 11;            // /2048
      const long nn = grow & 2047;
      dst[(((size_t)bb * NHEADS + h) * SEQ + nn) * HDIM + d] =
          (_Float16)(acc[t][r] * sc);
    }
  }
}

// ============================================================
// Kernel 2: fused flash attention per (b,h).
// 512 blocks x 256 threads. Block covers 128 query rows of one (b,h);
// wave w owns rows q0 = tile*128 + w*16. Key loop in tiles of 32.
// K tile staged via CDNA5 async global->LDS copy when available.
// ============================================================
__global__ __launch_bounds__(256) void attn_flash_kernel(
    const _Float16* __restrict__ Qh, const _Float16* __restrict__ Kh,
    const _Float16* __restrict__ Vh, const float* __restrict__ mask,
    float* __restrict__ Of) {
  __shared__ _Float16 sK[32 * 64];        // key tile, [key][d]
  __shared__ _Float16 sVT[64 * 32];       // value tile transposed, [d][key]
  __shared__ _Float16 sP[8 * 16 * 32];    // per-wave P staging, [wave][row][key]

  const int bh   = (int)blockIdx.x >> 4;       // 0..31
  const int tile = (int)blockIdx.x & 15;       // 0..15
  const int wave = (int)(threadIdx.x >> 5);
  const int lane = (int)(threadIdx.x & 31);
  const int l16  = lane & 15;
  const int rbase = (lane & 16) ? 8 : 0;
  const int b = bh >> 3, h = bh & 7;

  const _Float16* Qb = Qh + (size_t)bh * SEQ * HDIM;
  const _Float16* Kb = Kh + (size_t)bh * SEQ * HDIM;
  const _Float16* Vb = Vh + (size_t)bh * SEQ * HDIM;

  const int q0 = tile * 128 + wave * 16;

  // Q fragments (16 rows x 64 k) -> two 16x32 A fragments
  v16h qf0 = load_frag_h(Qb, q0 + l16, HDIM, 0, lane);
  v16h qf1 = load_frag_h(Qb, q0 + l16, HDIM, 32, lane);

  v8f oacc[4];
#pragma unroll
  for (int f = 0; f < 4; ++f)
#pragma unroll
    for (int i = 0; i < 8; ++i) oacc[f][i] = 0.0f;

  float mrun[8], lrun[8];
#pragma unroll
  for (int r = 0; r < 8; ++r) { mrun[r] = -3.0e38f; lrun[r] = 0.0f; }

  // staging indices for this thread (8 halves each for K and V)
  const int sidx = (int)threadIdx.x * 8;
  const int skey = sidx >> 6;
  const int sd   = sidx & 63;

  _Float16* Pw = sP + wave * (16 * 32);

  for (int j0 = 0; j0 < SEQ; j0 += 32) {
    __syncthreads();
    // ---- cooperative staging: K tile row-major, V tile transposed ----
    {
#if defined(USE_ASYNC_LDS)
      // CDNA5 async path: global -> LDS without touching VGPRs (ASYNCcnt)
      __builtin_amdgcn_global_load_async_to_lds_b128(
          to_glb(Kb + (size_t)(j0 + skey) * HDIM + sd),
          to_lds(&sK[skey * 64 + sd]), /*offset=*/0, /*cpol=*/0);
#else
      v8h kv = *(const v8h*)(Kb + (size_t)(j0 + skey) * HDIM + sd);
      *(v8h*)&sK[skey * 64 + sd] = kv;
#endif
      v8h vv = *(const v8h*)(Vb + (size_t)(j0 + skey) * HDIM + sd);
#pragma unroll
      for (int e = 0; e < 8; ++e) sVT[(sd + e) * 32 + skey] = vv[e];
      if (j0 + 32 < SEQ) {
        __builtin_prefetch(Kb + (size_t)(j0 + 32 + skey) * HDIM + sd, 0, 0);
        __builtin_prefetch(Vb + (size_t)(j0 + 32 + skey) * HDIM + sd, 0, 0);
      }
#if defined(USE_ASYNC_LDS)
      __builtin_amdgcn_s_wait_asynccnt(0);
#endif
    }
    __syncthreads();

    // ---- S = (Q*scale) @ K^T  : two 16x16 C fragments (keys c*16..c*16+15)
    v8f s[2];
#pragma unroll
    for (int c = 0; c < 2; ++c) {
#pragma unroll
      for (int i = 0; i < 8; ++i) s[c][i] = 0.0f;
      v16h kf0 = load_frag_h((const _Float16*)sK, c * 16 + l16, 64, 0, lane);
      v16h kf1 = load_frag_h((const _Float16*)sK, c * 16 + l16, 64, 32, lane);
      s[c] = wmma32(qf0, kf0, s[c]);
      s[c] = wmma32(qf1, kf1, s[c]);
    }

    // ---- add mask, online softmax over 32 keys ----
#pragma unroll
    for (int c = 0; c < 2; ++c)
#pragma unroll
      for (int r = 0; r < 8; ++r)
        s[c][r] += mask[(size_t)(q0 + rbase + r) * SEQ + (j0 + c * 16 + l16)];

    float rmax[8], rsum[8];
#pragma unroll
    for (int r = 0; r < 8; ++r) rmax[r] = fmaxf(s[0][r], s[1][r]);
#pragma unroll
    for (int off = 1; off < 16; off <<= 1)
#pragma unroll
      for (int r = 0; r < 8; ++r)
        rmax[r] = fmaxf(rmax[r], __shfl_xor(rmax[r], off, 32));

#pragma unroll
    for (int r = 0; r < 8; ++r) {
      const float mnew = fmaxf(mrun[r], rmax[r]);
      const float alpha = __expf(mrun[r] - mnew);
      mrun[r] = mnew;
      s[0][r] = __expf(s[0][r] - mnew);
      s[1][r] = __expf(s[1][r] - mnew);
      rsum[r] = s[0][r] + s[1][r];
      lrun[r] *= alpha;
#pragma unroll
      for (int f = 0; f < 4; ++f) oacc[f][r] *= alpha;
    }
#pragma unroll
    for (int off = 1; off < 16; off <<= 1)
#pragma unroll
      for (int r = 0; r < 8; ++r) rsum[r] += __shfl_xor(rsum[r], off, 32);
#pragma unroll
    for (int r = 0; r < 8; ++r) lrun[r] += rsum[r];

    // ---- P (C-layout) -> LDS -> A-layout fragment ----
#pragma unroll
    for (int c = 0; c < 2; ++c)
#pragma unroll
      for (int r = 0; r < 8; ++r)
        Pw[(rbase + r) * 32 + c * 16 + l16] = (_Float16)s[c][r];
    asm volatile("" ::: "memory");  // keep DS store->load order (HW in-order)
    v16h pf = load_frag_h((const _Float16*)Pw, l16, 32, 0, lane);

    // ---- O += P @ V : 4 output d-chunks, B frags from V^T (contiguous) ----
#pragma unroll
    for (int f = 0; f < 4; ++f) {
      v16h vf = load_frag_h((const _Float16*)sVT, f * 16 + l16, 32, 0, lane);
      oacc[f] = wmma32(pf, vf, oacc[f]);
    }
  }

  // ---- normalize and write O as [B, N, H*64+d] fp32 ----
#pragma unroll
  for (int r = 0; r < 8; ++r) {
    const float inv = 1.0f / lrun[r];
    const long n = q0 + rbase + r;
#pragma unroll
    for (int f = 0; f < 4; ++f)
      Of[((size_t)b * SEQ + n) * DIMC + h * HDIM + f * 16 + l16] =
          oacc[f][r] * inv;
  }
}

// ============================================================
// Kernel 3: output projection  out = O @ proj_w^T + proj_b  (fp32 out).
// One wave per 16x64 tile. M tiles = 512, N64 tiles = 8 -> 4096 waves.
// ============================================================
__global__ __launch_bounds__(256) void attn_proj_kernel(
    const float* __restrict__ Of, const float* __restrict__ proj_w,
    const float* __restrict__ proj_b, float* __restrict__ out) {
  const int gwave = (int)((blockIdx.x * blockDim.x + threadIdx.x) >> 5);
  const int lane  = (int)(threadIdx.x & 31);
  const int mt = gwave / 8;
  const int nt = gwave % 8;
  const int m0 = mt * 16;
  const int n0 = nt * 64;
  const int l16 = lane & 15;

  v8f acc[4];
#pragma unroll
  for (int t = 0; t < 4; ++t)
#pragma unroll
    for (int i = 0; i < 8; ++i) acc[t][i] = 0.0f;

#pragma unroll 2
  for (int k0 = 0; k0 < DIMC; k0 += 32) {
    v16h a = load_frag_f(Of, (long)(m0 + l16), DIMC, k0, lane);
#pragma unroll
    for (int t = 0; t < 4; ++t) {
      v16h b = load_frag_f(proj_w, (long)(n0 + t * 16 + l16), DIMC, k0, lane);
      acc[t] = wmma32(a, b, acc[t]);
    }
  }

  const int rbase = (lane & 16) ? 8 : 0;
#pragma unroll
  for (int t = 0; t < 4; ++t) {
    const int col = n0 + t * 16 + l16;
    const float bias = proj_b[col];
#pragma unroll
    for (int r = 0; r < 8; ++r) {
      const long grow = m0 + rbase + r;
      out[(size_t)grow * DIMC + col] = acc[t][r] + bias;
    }
  }
}

// ============================================================
extern "C" void kernel_launch(void* const* d_in, const int* in_sizes, int n_in,
                              void* d_out, int out_size, void* d_ws,
                              size_t ws_size, hipStream_t stream) {
  const float* x      = (const float*)d_in[0];   // [B,N,512]
  const float* mask   = (const float*)d_in[1];   // [1,1,N,N]
  const float* qkv_w  = (const float*)d_in[2];   // [1536,512]
  const float* proj_w = (const float*)d_in[3];   // [512,512]
  const float* proj_b = (const float*)d_in[4];   // [512]
  float* out = (float*)d_out;                    // [B,N,512]

  // workspace layout (f16 Q/K/V in [B,H,N,64], fp32 attention output)
  const size_t per = (size_t)BATCH * NHEADS * SEQ * HDIM;   // 4,194,304 elems
  _Float16* Qh = (_Float16*)d_ws;
  _Float16* Kh = Qh + per;
  _Float16* Vh = Kh + per;
  float*    Of = (float*)(Vh + per);            // B*N*512 fp32

  // 1) QKV projection: 512*24 = 12288 waves -> 1536 blocks of 256
  attn_qkv_kernel<<<1536, 256, 0, stream>>>(x, qkv_w, Qh, Kh, Vh);

  // 2) flash attention: (B*H)*(N/128) = 32*16 = 512 blocks of 256
  attn_flash_kernel<<<512, 256, 0, stream>>>(Qh, Kh, Vh, mask, Of);

  // 3) output projection: 512*8 = 4096 waves -> 512 blocks of 256
  attn_proj_kernel<<<512, 256, 0, stream>>>(Of, proj_w, proj_b, out);
}
